// MHA_50019189129217
// MI455X (gfx1250) — matrix operand order repaired and checked by reading, not compiled, not run on hardware
//
#include <hip/hip_runtime.h>

// ---------------------------------------------------------------------------
// MHA forward for MI455X (gfx1250, wave32, WMMA f32_16x16x32_f16,
// async global->LDS staging, double-buffered flash attention)
//   D=512, H=8, DH=64, B=4, L=2048
// ---------------------------------------------------------------------------

typedef __attribute__((ext_vector_type(16))) _Float16 v16h;
typedef __attribute__((ext_vector_type(8)))  _Float16 v8h;
typedef __attribute__((ext_vector_type(8)))  float    v8f;
typedef __attribute__((ext_vector_type(4)))  float    v4f;

#define D_MODEL 512
#define NHEAD   8
#define DHEAD   64
#define BATCH   4
#define SEQ     2048
#define MROWS   (BATCH * SEQ)          // 8192

// LDS row strides (halves) chosen so every 16-B access is 16-B aligned and
// the 16 lanes of a B-fragment read hit distinct banks.
#define KSTRIDE 88                      // 32-row K tile (keys x 64)
#define VSTRIDE 40                      // 64-row V^T tile (dh x 32)
#define PSTRIDE 48                      // per-wave 16x32 P tile

__device__ __forceinline__ v8f wmma16x16x32(v16h a, v16h b, v8f c) {
  return __builtin_amdgcn_wmma_f32_16x16x32_f16(false, a, false, b,
                                                (short)0, c, false, false);
}

// ---- fragment loaders ------------------------------------------------------
// A-fragment (16x32): lane = row M; K chunks [c0..c0+7],[c0+16..c0+23],
// c0 = (lane>=16)?8:0.
__device__ __forceinline__ v16h a_frag_f16(const _Float16* row, int c0) {
  v8h lo = *(const v8h*)(row + c0);
  v8h hh = *(const v8h*)(row + c0 + 16);
  v16h a;
#pragma unroll
  for (int j = 0; j < 8; ++j) { a[j] = lo[j]; a[j + 8] = hh[j]; }
  return a;
}

__device__ __forceinline__ v16h a_frag_f32(const float* row, int c0) {
  v4f x0 = *(const v4f*)(row + c0);
  v4f x1 = *(const v4f*)(row + c0 + 4);
  v4f x2 = *(const v4f*)(row + c0 + 16);
  v4f x3 = *(const v4f*)(row + c0 + 20);
  v16h a;
  {
    auto t = __builtin_amdgcn_cvt_pkrtz(x0[0], x0[1]);
    a[0] = (_Float16)t[0]; a[1] = (_Float16)t[1];
  }
  {
    auto t = __builtin_amdgcn_cvt_pkrtz(x0[2], x0[3]);
    a[2] = (_Float16)t[0]; a[3] = (_Float16)t[1];
  }
  {
    auto t = __builtin_amdgcn_cvt_pkrtz(x1[0], x1[1]);
    a[4] = (_Float16)t[0]; a[5] = (_Float16)t[1];
  }
  {
    auto t = __builtin_amdgcn_cvt_pkrtz(x1[2], x1[3]);
    a[6] = (_Float16)t[0]; a[7] = (_Float16)t[1];
  }
  {
    auto t = __builtin_amdgcn_cvt_pkrtz(x2[0], x2[1]);
    a[8] = (_Float16)t[0]; a[9] = (_Float16)t[1];
  }
  {
    auto t = __builtin_amdgcn_cvt_pkrtz(x2[2], x2[3]);
    a[10] = (_Float16)t[0]; a[11] = (_Float16)t[1];
  }
  {
    auto t = __builtin_amdgcn_cvt_pkrtz(x3[0], x3[1]);
    a[12] = (_Float16)t[0]; a[13] = (_Float16)t[1];
  }
  {
    auto t = __builtin_amdgcn_cvt_pkrtz(x3[2], x3[3]);
    a[14] = (_Float16)t[0]; a[15] = (_Float16)t[1];
  }
  return a;
}

// B-fragment (32x16): lane = column N; K chunks [half*8..+7],[16+half*8..+7].
__device__ __forceinline__ v16h b_frag_f16(const _Float16* row, int half) {
  v8h lo = *(const v8h*)(row + half * 8);
  v8h hh = *(const v8h*)(row + half * 8 + 16);
  v16h b;
#pragma unroll
  for (int j = 0; j < 8; ++j) { b[j] = lo[j]; b[j + 8] = hh[j]; }
  return b;
}

// ---- CDNA5 async global -> LDS copy (ASYNCcnt) -----------------------------
__device__ __forceinline__ void async_copy_b128(unsigned lds_off,
                                                const void* gaddr) {
  asm volatile("global_load_async_to_lds_b128 %0, %1, off"
               :: "v"(lds_off), "v"(gaddr) : "memory");
}
__device__ __forceinline__ void wait_async0() {
  asm volatile("s_wait_asynccnt 0x0" ::: "memory");
}
__device__ __forceinline__ void wait_ds0() {
  asm volatile("s_wait_dscnt 0x0" ::: "memory");
}

// ---------------------------------------------------------------------------
// One-time f32 -> f16 weight conversion (packed cvt); n = 512*512, 8/thread.
// ---------------------------------------------------------------------------
__global__ void __launch_bounds__(256)
cvt_w_kernel(const float* __restrict__ src, _Float16* __restrict__ dst) {
  const int i = (blockIdx.x * 256 + threadIdx.x) * 8;
  v4f a = *(const v4f*)(src + i);
  v4f b = *(const v4f*)(src + i + 4);
  v8h o;
  {
    auto t = __builtin_amdgcn_cvt_pkrtz(a[0], a[1]);
    o[0] = (_Float16)t[0]; o[1] = (_Float16)t[1];
  }
  {
    auto t = __builtin_amdgcn_cvt_pkrtz(a[2], a[3]);
    o[2] = (_Float16)t[0]; o[3] = (_Float16)t[1];
  }
  {
    auto t = __builtin_amdgcn_cvt_pkrtz(b[0], b[1]);
    o[4] = (_Float16)t[0]; o[5] = (_Float16)t[1];
  }
  {
    auto t = __builtin_amdgcn_cvt_pkrtz(b[2], b[3]);
    o[6] = (_Float16)t[0]; o[7] = (_Float16)t[1];
  }
  *(v8h*)(dst + i) = o;
}

// ---------------------------------------------------------------------------
// Y = X @ W^T + bias  (X: 8192x512, W16: f16 512x512 row-major)
// MODE 0: X f32 -> f16 per-head row-major  qk[bh][l][p]
// MODE 1: X f32 -> f16 per-head TRANSPOSED vT[bh][p][l]
// MODE 2: X f16 -> f32 Y[m][n]
// Block = 256 thr = 8 waves; wave tile 16(M) x 64(N); grid = (64, 8)
// ---------------------------------------------------------------------------
template <int MODE>
__global__ void __launch_bounds__(256)
gemm_xWt(const void* __restrict__ Xv, const _Float16* __restrict__ W16,
         const float* __restrict__ bias, void* __restrict__ Yv) {
  const int lane = threadIdx.x & 31;
  const int wave = threadIdx.x >> 5;
  const int hi   = lane >> 4;
  const int c0   = hi ? 8 : 0;
  const int col  = lane & 15;
  const int m0   = blockIdx.x * 128 + wave * 16;
  const int n0   = blockIdx.y * 64;
  const int mrow = m0 + col;

  v8f acc[4] = {};

#pragma unroll 2
  for (int kt = 0; kt < D_MODEL; kt += 32) {
    v16h a;
    if (MODE == 2)
      a = a_frag_f16((const _Float16*)Xv + mrow * D_MODEL + kt, c0);
    else
      a = a_frag_f32((const float*)Xv + mrow * D_MODEL + kt, c0);
    // hoist all four weight fragments, then issue the four WMMAs
    v16h bfr[4];
#pragma unroll
    for (int nf = 0; nf < 4; ++nf) {
      const int n = n0 + nf * 16 + col;
      bfr[nf] = b_frag_f16(W16 + n * D_MODEL + kt, hi);
    }
#pragma unroll
    for (int nf = 0; nf < 4; ++nf)
      acc[nf] = wmma16x16x32(a, bfr[nf], acc[nf]);
  }

#pragma unroll
  for (int nf = 0; nf < 4; ++nf) {
    const int n  = n0 + nf * 16 + col;
    const float bn = bias[n];
    if (MODE == 2) {
      float* Y = (float*)Yv;
#pragma unroll
      for (int r = 0; r < 8; ++r) {
        const int m = m0 + hi * 8 + r;
        Y[m * D_MODEL + n] = acc[nf][r] + bn;
      }
    } else {
      _Float16* Y = (_Float16*)Yv;
      const int h = n >> 6, p = n & 63;
      if (MODE == 0) {
#pragma unroll
        for (int r = 0; r < 8; ++r) {
          const int m = m0 + hi * 8 + r;
          const int bb = m >> 11, l = m & 2047;
          Y[((bb * NHEAD + h) * SEQ + l) * DHEAD + p] =
              (_Float16)(acc[nf][r] + bn);
        }
      } else {  // transposed: 8 consecutive l per lane -> one 16-B store
        const int m = m0 + hi * 8;
        const int bb = m >> 11, l = m & 2047;
        v8h pk;
#pragma unroll
        for (int r = 0; r < 8; ++r) pk[r] = (_Float16)(acc[nf][r] + bn);
        *(v8h*)&Y[((bb * NHEAD + h) * DHEAD + p) * SEQ + l] = pk;
      }
    }
  }
}

// ---------------------------------------------------------------------------
// Flash attention per (b,h). Block = 8 waves = 128 queries; wave owns 16 rows.
// K/V tiles (32 keys) staged to LDS with async copies, double-buffered.
// grid = (SEQ/128, B*H) = (16, 32)
// ---------------------------------------------------------------------------
__global__ void __launch_bounds__(256)
attn_kernel(const _Float16* __restrict__ qh, const _Float16* __restrict__ kh,
            const _Float16* __restrict__ vT, _Float16* __restrict__ ctx) {
  __shared__ _Float16 kbuf[2][32 * KSTRIDE];   // keys x dh   (11264 B)
  __shared__ _Float16 vbuf[2][64 * VSTRIDE];   // dh x keys   (10240 B)
  __shared__ _Float16 pbuf[8][16 * PSTRIDE];   // per-wave P  (12288 B)

  const int lane = threadIdx.x & 31;
  const int wave = threadIdx.x >> 5;
  const int hi   = lane >> 4;
  const int c0   = hi ? 8 : 0;
  const int col  = lane & 15;
  const int bh   = blockIdx.y;
  const int b    = bh >> 3;
  const int h    = bh & 7;
  const int q0   = blockIdx.x * 128 + wave * 16;

  // cooperative tile-loader coordinates (256 x 16B covers each 4KB tile)
  const int tkr = threadIdx.x >> 3, tkc = (threadIdx.x & 7) * 8;
  const int tvr = threadIdx.x >> 2, tvc = (threadIdx.x & 3) * 8;
  const _Float16* kg = kh + (bh * SEQ + tkr) * DHEAD + tkc;   // advance by kt*DHEAD
  const _Float16* vg = vT + (bh * DHEAD + tvr) * SEQ + tvc;   // advance by kt
  const unsigned kl0 = (unsigned)(size_t)&kbuf[0][tkr * KSTRIDE + tkc];
  const unsigned kl1 = (unsigned)(size_t)&kbuf[1][tkr * KSTRIDE + tkc];
  const unsigned vl0 = (unsigned)(size_t)&vbuf[0][tvr * VSTRIDE + tvc];
  const unsigned vl1 = (unsigned)(size_t)&vbuf[1][tvr * VSTRIDE + tvc];

  // Q fragments, pre-scaled by 1/sqrt(DH)=0.125 (exact in f16)
  const _Float16* qrow = qh + (bh * SEQ + q0 + col) * DHEAD;
  v16h aq0 = a_frag_f16(qrow, c0);
  v16h aq1 = a_frag_f16(qrow + 32, c0);
  const _Float16 qs = (_Float16)0.125f;
#pragma unroll
  for (int j = 0; j < 16; ++j) { aq0[j] *= qs; aq1[j] *= qs; }

  v8f o[4] = {};
  float mrow[8], lrow[8];
#pragma unroll
  for (int r = 0; r < 8; ++r) { mrow[r] = -3.0e38f; lrow[r] = 0.0f; }

  _Float16* myl = pbuf[wave];

  // prologue: stage tile 0
  async_copy_b128(kl0, kg);
  async_copy_b128(vl0, vg);
  wait_async0();
  __syncthreads();

  for (int kt = 0; kt < SEQ; kt += 32) {
    const int  cur  = (kt >> 5) & 1;
    const bool more = (kt + 32) < SEQ;
    if (more) {  // prefetch next tile into the other buffer (async)
      async_copy_b128(cur ? kl0 : kl1, kg + (kt + 32) * DHEAD);
      async_copy_b128(cur ? vl0 : vl1, vg + (kt + 32));
    }
    const _Float16* kb = kbuf[cur];
    const _Float16* vb = vbuf[cur];

    // ---- S = (Q*scale) @ K^T for 32 keys: load all 4 K-fragments first ----
    v16h kf0 = b_frag_f16(kb + col * KSTRIDE, hi);
    v16h kf1 = b_frag_f16(kb + col * KSTRIDE + 32, hi);
    v16h kf2 = b_frag_f16(kb + (16 + col) * KSTRIDE, hi);
    v16h kf3 = b_frag_f16(kb + (16 + col) * KSTRIDE + 32, hi);
    v8f s0 = {}, s1 = {};
    s0 = wmma16x16x32(aq0, kf0, s0);
    s0 = wmma16x16x32(aq1, kf1, s0);
    s1 = wmma16x16x32(aq0, kf2, s1);
    s1 = wmma16x16x32(aq1, kf3, s1);

    // issue V-fragment loads now so their LDS latency hides under softmax
    v16h vf[4];
#pragma unroll
    for (int nf = 0; nf < 4; ++nf)
      vf[nf] = b_frag_f16(vb + (nf * 16 + col) * VSTRIDE, hi);

    // ---- online softmax (rows live in half-waves of 16 lanes) ----
    float av[8];
#pragma unroll
    for (int r = 0; r < 8; ++r) {
      float t = fmaxf(s0[r], s1[r]);
      t = fmaxf(t, __shfl_xor(t, 8, 16));
      t = fmaxf(t, __shfl_xor(t, 4, 16));
      t = fmaxf(t, __shfl_xor(t, 2, 16));
      t = fmaxf(t, __shfl_xor(t, 1, 16));
      const float mo = mrow[r];
      const float mn = fmaxf(mo, t);
      const float al = __expf(mo - mn);
      mrow[r] = mn;
      const float p0 = __expf(s0[r] - mn);
      const float p1 = __expf(s1[r] - mn);
      float rs = p0 + p1;
      rs += __shfl_xor(rs, 8, 16);
      rs += __shfl_xor(rs, 4, 16);
      rs += __shfl_xor(rs, 2, 16);
      rs += __shfl_xor(rs, 1, 16);
      lrow[r] = lrow[r] * al + rs;
      av[r] = al;
      myl[(hi * 8 + r) * PSTRIDE + col]      = (_Float16)p0;
      myl[(hi * 8 + r) * PSTRIDE + 16 + col] = (_Float16)p1;
    }
    wait_ds0();

    // reload P as an A-fragment (wave-private LDS, no barrier needed)
    v16h pa;
    {
      const _Float16* lp = myl + col * PSTRIDE;
      v8h lo = *(const v8h*)(lp + c0);
      v8h hh = *(const v8h*)(lp + c0 + 16);
#pragma unroll
      for (int j = 0; j < 8; ++j) { pa[j] = lo[j]; pa[j + 8] = hh[j]; }
    }

    // ---- O = O*alpha + P @ V ----
#pragma unroll
    for (int nf = 0; nf < 4; ++nf) {
#pragma unroll
      for (int r = 0; r < 8; ++r) o[nf][r] *= av[r];
      o[nf] = wmma16x16x32(pa, vf[nf], o[nf]);
    }

    if (more) wait_async0();   // next tile landed (this wave's copies)
    __syncthreads();           // publish; also closes read of `cur`
  }

  // ---- normalize, write ctx (f16, [b][l][h*64+n]) ----
  float inv[8];
#pragma unroll
  for (int r = 0; r < 8; ++r) inv[r] = 1.0f / lrow[r];
#pragma unroll
  for (int nf = 0; nf < 4; ++nf) {
    const int n = nf * 16 + col;
#pragma unroll
    for (int r = 0; r < 8; ++r) {
      const int m = q0 + hi * 8 + r;
      ctx[(b * SEQ + m) * D_MODEL + h * DHEAD + n] =
          (_Float16)(o[nf][r] * inv[r]);
    }
  }
}

// ---------------------------------------------------------------------------
extern "C" void kernel_launch(void* const* d_in, const int* in_sizes, int n_in,
                              void* d_out, int out_size, void* d_ws,
                              size_t ws_size, hipStream_t stream) {
  (void)in_sizes; (void)n_in; (void)out_size; (void)ws_size;

  const float* Q  = (const float*)d_in[0];
  const float* K  = (const float*)d_in[1];
  const float* V  = (const float*)d_in[2];
  const float* Wq = (const float*)d_in[3];
  const float* bq = (const float*)d_in[4];
  const float* Wk = (const float*)d_in[5];
  const float* bk = (const float*)d_in[6];
  const float* Wv = (const float*)d_in[7];
  const float* bv = (const float*)d_in[8];
  const float* Wo = (const float*)d_in[9];
  const float* bo = (const float*)d_in[10];

  const size_t headElems = (size_t)BATCH * NHEAD * SEQ * DHEAD;  // 4 Mi
  const size_t wElems    = (size_t)D_MODEL * D_MODEL;            // 256 Ki
  _Float16* qh   = (_Float16*)d_ws;
  _Float16* khp  = qh + headElems;
  _Float16* vTp  = khp + headElems;
  _Float16* ctx  = vTp + headElems;
  _Float16* wq16 = ctx + headElems;
  _Float16* wk16 = wq16 + wElems;
  _Float16* wv16 = wk16 + wElems;
  _Float16* wo16 = wv16 + wElems;

  const dim3 blk(256);
  const dim3 gCvt(wElems / (256 * 8));           // 128
  const dim3 gGemm(MROWS / 128, D_MODEL / 64);   // (64, 8)
  const dim3 gAttn(SEQ / 128, BATCH * NHEAD);    // (16, 32)

  cvt_w_kernel<<<gCvt, blk, 0, stream>>>(Wq, wq16);
  cvt_w_kernel<<<gCvt, blk, 0, stream>>>(Wk, wk16);
  cvt_w_kernel<<<gCvt, blk, 0, stream>>>(Wv, wv16);
  cvt_w_kernel<<<gCvt, blk, 0, stream>>>(Wo, wo16);

  gemm_xWt<0><<<gGemm, blk, 0, stream>>>(Q, wq16, bq, qh);
  gemm_xWt<0><<<gGemm, blk, 0, stream>>>(K, wk16, bk, khp);
  gemm_xWt<1><<<gGemm, blk, 0, stream>>>(V, wv16, bv, vTp);
  attn_kernel<<<gAttn, blk, 0, stream>>>(qh, khp, vTp, ctx);
  gemm_xWt<2><<<gGemm, blk, 0, stream>>>(ctx, wo16, bo, d_out);
}
